// Non_Local_Block_80788334838324
// MI455X (gfx1250) — compile-verified, matrix operand order checked by reading, not run on hardware
//
#include <hip/hip_runtime.h>
#include <hip/hip_bf16.h>
#include <math.h>

// ---------------------------------------------------------------------------
// Problem constants (from reference)
// ---------------------------------------------------------------------------
constexpr int B_  = 4;
constexpr int C_  = 256;
constexpr int Ci_ = 128;
constexpr int T_  = 8;
constexpr int H_  = 32;
constexpr int W_  = 32;
constexpr int N_  = T_ * H_ * W_;        // 8192
constexpr int H2_ = 16;
constexpr int W2_ = 16;
constexpr int N2_ = T_ * H2_ * W2_;      // 2048

// ---------------------------------------------------------------------------
// WMMA types (CDNA5 wave32): bf16 A/B (16 elems/lane), f32 C/D (8 elems/lane)
// ---------------------------------------------------------------------------
typedef __attribute__((ext_vector_type(16))) __bf16          v16bf;
typedef __attribute__((ext_vector_type(8)))  float           v8f;

union Frag16 {
    v16bf v;
    unsigned short u[16];
};

__device__ __forceinline__ v8f wmma_bf16(const v16bf& a, const v16bf& b, const v8f& c) {
    // D = A(16x32) * B(32x16) + C(16x16), f32 accumulate
    return __builtin_amdgcn_wmma_f32_16x16x32_bf16(
        /*neg_a=*/false, a, /*neg_b=*/false, b,
        /*c_mod=*/(short)0, c, /*reuse_a=*/false, /*reuse_b=*/false);
}

// f32 -> bf16 bits, round-to-nearest-even
__device__ __forceinline__ unsigned short f2bf(float x) {
    unsigned int u = __float_as_uint(x);
    unsigned int r = (u + 0x7FFFu + ((u >> 16) & 1u)) >> 16;
    return (unsigned short)r;
}
__device__ __forceinline__ float bf2f(unsigned short b) {
    return __uint_as_float(((unsigned int)b) << 16);
}

// A/B operand K striping within a lane (16-bit elements, 16x32 tile):
//  element e (0..15): k = k0 + e + (e>=8 ? 8 : 0) + half*8
__device__ __forceinline__ int frag_k(int k0, int e, int hl) {
    return k0 + e + ((e >= 8) ? 8 : 0) + hl * 8;
}

// Load a K-contiguous 16x32 operand fragment: element (row, k) = base[row*ld + k]
// "row" is M for A operands, N for B operands (both use lane%16 as the row).
__device__ __forceinline__ void load_frag_rowmajor(Frag16& f, const unsigned short* row,
                                                   int k0, int hl) {
#pragma unroll
    for (int j = 0; j < 8; ++j) {
        const int e = 2 * j;
        const int k = frag_k(k0, e, hl);
        f.u[e]     = row[k];
        f.u[e + 1] = row[k + 1];
    }
}

// Async global -> LDS copy of 16 bytes per lane (CDNA5 VGLOBAL op 98),
// tracked by ASYNCcnt. ldst = LDS byte offset (low 32 bits of generic addr).
__device__ __forceinline__ void async_copy_b128(unsigned lds_off,
                                                const unsigned short* gsrc) {
    asm volatile("global_load_async_to_lds_b128 %0, %1, off"
                 :: "v"(lds_off), "v"(gsrc) : "memory");
}

// ---------------------------------------------------------------------------
// Kernel 0a: f32 -> bf16 bulk convert (weights)
// ---------------------------------------------------------------------------
__global__ void cvt_f32_bf16_kernel(const float* __restrict__ src,
                                    unsigned short* __restrict__ dst, int n) {
    int i = blockIdx.x * blockDim.x + threadIdx.x;
    if (i < n) dst[i] = f2bf(src[i]);
}

// ---------------------------------------------------------------------------
// Kernel 0b: x [B, C, N] f32 -> xt [B, N, C] bf16, tiled LDS transpose.
// grid (N/16, C/16, B), block (16, 16).
// ---------------------------------------------------------------------------
__global__ void transpose_cvt_kernel(const float* __restrict__ x,
                                     unsigned short* __restrict__ xt) {
    __shared__ float tile[16][17];
    const int b  = blockIdx.z;
    const int n0 = blockIdx.x * 16;
    const int c0 = blockIdx.y * 16;
    const int tx = threadIdx.x;
    const int ty = threadIdx.y;

    tile[ty][tx] = x[((size_t)b * C_ + c0 + ty) * N_ + n0 + tx];
    __syncthreads();
    xt[((size_t)b * N_ + n0 + ty) * C_ + c0 + tx] = f2bf(tile[tx][ty]);
}

// ---------------------------------------------------------------------------
// Kernel 1: conv1x1 as GEMM in [spatial, channel] orientation.
//   outT[n, ci] = xt[n, :] . w[ci, :] + bias[ci]
// Block: 256 threads = 8 waves; wave w owns ci-tile w*16; grid (N/16, B).
// ---------------------------------------------------------------------------
__global__ void conv1x1_gemm_kernel(const unsigned short* __restrict__ xt,
                                    const unsigned short* __restrict__ wb,
                                    const float* __restrict__ bias,
                                    unsigned short* __restrict__ outT) {
    const int b    = blockIdx.y;
    const int wave = threadIdx.x >> 5;
    const int lane = threadIdx.x & 31;
    const int hl   = lane >> 4;
    const int ln   = lane & 15;
    const int n0   = blockIdx.x * 16;
    const int ci0  = wave * 16;

    const unsigned short* arow = xt + ((size_t)b * N_ + n0 + ln) * C_;
    const unsigned short* brow = wb + (size_t)(ci0 + ln) * C_;

    v8f acc;
    const float bv = bias[ci0 + ln];     // bias varies along the N (=ci) dim
#pragma unroll
    for (int r = 0; r < 8; ++r) acc[r] = bv;

#pragma unroll
    for (int s = 0; s < C_ / 32; ++s) {  // 8 K-steps of 32
        Frag16 aX, bW;
        load_frag_rowmajor(aX, arow, s * 32, hl);
        load_frag_rowmajor(bW, brow, s * 32, hl);
        acc = wmma_bf16(aX.v, bW.v, acc);
    }

#pragma unroll
    for (int r = 0; r < 8; ++r) {
        const int m = r + hl * 8;
        outT[((size_t)b * N_ + n0 + m) * Ci_ + ci0 + ln] = f2bf(acc[r]);
    }
}

// ---------------------------------------------------------------------------
// Kernel 2: fused MaxPool3d (1,2,2) for phi and g.
//   phi_f/g_f [B, N, Ci] -> phi_p [B, N2, Ci]  (key-major, for Q.K staging)
//                        -> gT    [B, Ci, N2]  (ci-major, for async PV staging)
// ---------------------------------------------------------------------------
__global__ void pool_1x2x2_kernel(const unsigned short* __restrict__ phi_f,
                                  const unsigned short* __restrict__ g_f,
                                  unsigned short* __restrict__ phi_p,
                                  unsigned short* __restrict__ gT) {
    const int total = B_ * N2_ * Ci_;
    int idx = blockIdx.x * blockDim.x + threadIdx.x;
    if (idx >= total) return;
    const int ci = idx & (Ci_ - 1);
    const int n2 = (idx >> 7) & (N2_ - 1);
    const int b  = idx >> 18;                       // / (N2*Ci)
    const int w2 = n2 & 15;
    const int h2 = (n2 >> 4) & 15;
    const int t  = n2 >> 8;
    const size_t base =
        ((size_t)b * N_ + t * (H_ * W_) + (h2 * 2) * W_ + w2 * 2) * Ci_ + ci;

    const unsigned short* p = phi_f + base;
    const float vp = fmaxf(fmaxf(bf2f(p[0]), bf2f(p[(size_t)Ci_])),
                           fmaxf(bf2f(p[(size_t)W_ * Ci_]),
                                 bf2f(p[(size_t)(W_ + 1) * Ci_])));
    phi_p[idx] = f2bf(vp);

    const unsigned short* q = g_f + base;
    const float vg = fmaxf(fmaxf(bf2f(q[0]), bf2f(q[(size_t)Ci_])),
                           fmaxf(bf2f(q[(size_t)W_ * Ci_]),
                                 bf2f(q[(size_t)(W_ + 1) * Ci_])));
    gT[((size_t)b * Ci_ + ci) * N2_ + n2] = f2bf(vg);
}

// ---------------------------------------------------------------------------
// Kernel 3: flash attention with double-buffered async global->LDS staging.
//   theta_t : [B, N,  Ci] bf16 (queries)
//   phi_p   : [B, N2, Ci] bf16 (keys, key-major)
//   gT      : [B, Ci, N2] bf16 (values, ci-major)
//   y_out   : [B, N,  Ci] bf16
// Block = 128 threads = 4 waves; wave owns one 16-query tile.
// Keys in chunks of 32: while computing on buffer `cur`, chunk kc+1 streams
// into buffer `cur^1` via GLOBAL_LOAD_ASYNC_TO_LDS_B128 (8 async instructions
// per wave per chunk -> s_wait_asynccnt 0x8 completes the older chunk).
// ---------------------------------------------------------------------------
__global__ void __launch_bounds__(128)
attention_kernel(const unsigned short* __restrict__ theta_t,
                 const unsigned short* __restrict__ phi_p,
                 const unsigned short* __restrict__ gT,
                 unsigned short* __restrict__ y_out) {
    constexpr int KC  = 32;            // keys per chunk
    constexpr int LDP = Ci_ + 8;       // 136 u16 = 272 B rows (16B aligned)
    constexpr int LDK = KC + 8;        // 40 u16 = 80 B rows (16B aligned)
    constexpr int NW  = 4;             // waves per block

    __shared__ unsigned short s_phi[2][KC][LDP];    // [buf][key][ci]
    __shared__ unsigned short s_g[2][Ci_][LDK];     // [buf][ci][key]
    __shared__ unsigned short s_p[NW][16][KC + 2];  // per-wave P as [m][key]

    const int b    = blockIdx.y;
    const int wave = threadIdx.x >> 5;
    const int lane = threadIdx.x & 31;
    const int hl   = lane >> 4;
    const int ln   = lane & 15;
    const int q0   = (blockIdx.x * NW + wave) * 16;

    const unsigned short* th = theta_t + (size_t)b * N_ * Ci_;
    const unsigned short* ph = phi_p   + (size_t)b * N2_ * Ci_;
    const unsigned short* gg = gT      + (size_t)b * Ci_ * N2_;

    // Issue one chunk's async copies: phi [32key x 128ci] and g [128ci x 32key].
    auto issue_chunk = [&](int kc, int buf) {
#pragma unroll
        for (int it = 0; it < 4; ++it) {            // phi: 512 x b128 / block
            const int idx = (int)threadIdx.x + it * 128;
            const int key = idx >> 4;
            const int cv  = (idx & 15) * 8;
            async_copy_b128((unsigned)(size_t)&s_phi[buf][key][cv],
                            ph + ((size_t)kc * KC + key) * Ci_ + cv);
        }
#pragma unroll
        for (int it = 0; it < 4; ++it) {            // g: 512 x b128 / block
            const int idx = (int)threadIdx.x + it * 128;
            const int ci  = idx >> 2;
            const int kk  = (idx & 3) * 8;
            async_copy_b128((unsigned)(size_t)&s_g[buf][ci][kk],
                            gg + (size_t)ci * N2_ + (size_t)kc * KC + kk);
        }
    };

    // Load theta A-fragments once (K = Ci = 128 -> 4 fragments of 32)
    Frag16 aTh[4];
    {
        const unsigned short* row = th + (size_t)(q0 + ln) * Ci_;
#pragma unroll
        for (int s = 0; s < 4; ++s) load_frag_rowmajor(aTh[s], row, s * 32, hl);
    }

    const v8f vz = {};
    v8f accY[8];
#pragma unroll
    for (int i = 0; i < 8; ++i) accY[i] = vz;

    float rowM[8], rowL[8];
#pragma unroll
    for (int r = 0; r < 8; ++r) { rowM[r] = -INFINITY; rowL[r] = 0.0f; }

    const int nChunks = N2_ / KC;      // 64
    issue_chunk(0, 0);                 // prime the pipeline

    for (int kc = 0; kc < nChunks; ++kc) {
        const int cur = kc & 1;
        if (kc + 1 < nChunks) {
            issue_chunk(kc + 1, cur ^ 1);
            // 8 newer async ops outstanding; <=8 means chunk kc has landed
            asm volatile("s_wait_asynccnt 0x8" ::: "memory");
        } else {
            asm volatile("s_wait_asynccnt 0x0" ::: "memory");
        }
        __syncthreads();               // chunk kc visible to all waves

        // --- GEMM1: S[16q x 32k] = theta_tile @ phi_chunk (K = 128) ---
        v8f S[2];
        S[0] = vz; S[1] = vz;
#pragma unroll
        for (int j = 0; j < 2; ++j) {
            const unsigned short* col = &s_phi[cur][j * 16 + ln][0];  // B(k=ci,n=key)
#pragma unroll
            for (int s = 0; s < 4; ++s) {
                Frag16 bPh;
                load_frag_rowmajor(bPh, col, s * 32, hl);
                S[j] = wmma_bf16(aTh[s].v, bPh.v, S[j]);
            }
        }

        // --- online softmax over this 32-key chunk ---
#pragma unroll
        for (int r = 0; r < 8; ++r) {
            float cm = fmaxf(S[0][r], S[1][r]);
#pragma unroll
            for (int msk = 8; msk >= 1; msk >>= 1)
                cm = fmaxf(cm, __shfl_xor(cm, msk, 32));    // 16-lane butterfly

            const float mOld = rowM[r];
            const float mNew = fmaxf(mOld, cm);
            const float scl  = __expf(mOld - mNew);
            rowM[r] = mNew;

            const float p0 = __expf(S[0][r] - mNew);
            const float p1 = __expf(S[1][r] - mNew);
            float ls = p0 + p1;
#pragma unroll
            for (int msk = 8; msk >= 1; msk >>= 1)
                ls += __shfl_xor(ls, msk, 32);
            rowL[r] = rowL[r] * scl + ls;

#pragma unroll
            for (int cg = 0; cg < 8; ++cg) accY[cg][r] *= scl;

            // write P to LDS in [m][key] order for the A-operand of GEMM2
            const int m = r + hl * 8;
            s_p[wave][m][ln]      = f2bf(p0);
            s_p[wave][m][16 + ln] = f2bf(p1);
        }
        __syncthreads();   // order P writes vs. cross-lane reads

        // --- GEMM2: Y[16q x 128ci] += P[16q x 32k] @ g_chunk^T[32k x 128ci] ---
        Frag16 aP;
        load_frag_rowmajor(aP, &s_p[wave][ln][0], 0, hl);
#pragma unroll
        for (int cg = 0; cg < 8; ++cg) {
            Frag16 bG;     // B(k=key, n=ci) = s_g[ci][key], K-contiguous
            load_frag_rowmajor(bG, &s_g[cur][cg * 16 + ln][0], 0, hl);
            accY[cg] = wmma_bf16(aP.v, bG.v, accY[cg]);
        }
        __syncthreads();   // all waves done with buffer `cur` before reuse
    }

    // --- normalize and store y [B, N, Ci] ---
#pragma unroll
    for (int r = 0; r < 8; ++r) {
        const float inv = 1.0f / rowL[r];
        const int m = r + hl * 8;
#pragma unroll
        for (int cg = 0; cg < 8; ++cg) {
            y_out[((size_t)b * N_ + q0 + m) * Ci_ + cg * 16 + ln] =
                f2bf(accY[cg][r] * inv);
        }
    }
}

// ---------------------------------------------------------------------------
// Kernel 4: final conv1x1 + BatchNorm(eval) + residual
// ---------------------------------------------------------------------------
__global__ void final_gemm_kernel(const unsigned short* __restrict__ yb,
                                  const unsigned short* __restrict__ wf,
                                  const float* __restrict__ b_final,
                                  const float* __restrict__ gamma,
                                  const float* __restrict__ beta,
                                  const float* __restrict__ mean,
                                  const float* __restrict__ var,
                                  const float* __restrict__ x,
                                  float* __restrict__ out) {
    const int b    = blockIdx.y;
    const int wave = threadIdx.x >> 5;
    const int lane = threadIdx.x & 31;
    const int hl   = lane >> 4;
    const int ln   = lane & 15;
    const int n0   = blockIdx.x * 16;

    const unsigned short* yrow = yb + ((size_t)b * N_ + n0 + ln) * Ci_;

#pragma unroll
    for (int cb = 0; cb < 2; ++cb) {
        const int c0 = (cb * 8 + wave) * 16;
        const unsigned short* wrow = wf + (size_t)(c0 + ln) * Ci_;

        v8f acc = {};
#pragma unroll
        for (int s = 0; s < Ci_ / 32; ++s) {   // 4 K-steps
            Frag16 aW, bY;
            load_frag_rowmajor(aW, wrow, s * 32, hl);
            load_frag_rowmajor(bY, yrow, s * 32, hl);
            acc = wmma_bf16(aW.v, bY.v, acc);
        }

#pragma unroll
        for (int r = 0; r < 8; ++r) {
            const int c = c0 + r + hl * 8;
            const float inv = gamma[c] * rsqrtf(var[c] + 1e-5f);
            const float val = (acc[r] + b_final[c] - mean[c]) * inv + beta[c];
            const size_t oi = ((size_t)b * C_ + c) * N_ + n0 + ln;
            out[oi] = val + x[oi];
        }
    }
}

// ---------------------------------------------------------------------------
// Host launcher
// ---------------------------------------------------------------------------
extern "C" void kernel_launch(void* const* d_in, const int* in_sizes, int n_in,
                              void* d_out, int out_size, void* d_ws, size_t ws_size,
                              hipStream_t stream) {
    (void)in_sizes; (void)n_in; (void)out_size; (void)ws_size;

    const float* x        = (const float*)d_in[0];
    const float* w_theta  = (const float*)d_in[1];
    const float* b_theta  = (const float*)d_in[2];
    const float* w_phi    = (const float*)d_in[3];
    const float* b_phi    = (const float*)d_in[4];
    const float* w_g      = (const float*)d_in[5];
    const float* b_g      = (const float*)d_in[6];
    const float* w_final  = (const float*)d_in[7];
    const float* b_final  = (const float*)d_in[8];
    const float* bn_gamma = (const float*)d_in[9];
    const float* bn_beta  = (const float*)d_in[10];
    const float* bn_mean  = (const float*)d_in[11];
    const float* bn_var   = (const float*)d_in[12];
    float* out = (float*)d_out;

    // Workspace carve-up (bf16 buffers), 256-byte aligned
    unsigned char* ws = (unsigned char*)d_ws;
    auto carve = [&ws](size_t bytes) -> unsigned short* {
        unsigned short* p = (unsigned short*)ws;
        ws += (bytes + 255) & ~(size_t)255;
        return p;
    };
    unsigned short* xt      = carve((size_t)B_ * N_ * C_ * 2);   // [B,N,C]
    unsigned short* wq      = carve((size_t)Ci_ * C_ * 2);
    unsigned short* wk      = carve((size_t)Ci_ * C_ * 2);
    unsigned short* wv      = carve((size_t)Ci_ * C_ * 2);
    unsigned short* wfb     = carve((size_t)C_ * Ci_ * 2);
    unsigned short* theta_t = carve((size_t)B_ * N_ * Ci_ * 2);  // [B,N,Ci]
    unsigned short* phi_f   = carve((size_t)B_ * N_ * Ci_ * 2);  // [B,N,Ci]
    unsigned short* g_f     = carve((size_t)B_ * N_ * Ci_ * 2);  // [B,N,Ci]
    unsigned short* phi_p   = carve((size_t)B_ * N2_ * Ci_ * 2); // [B,N2,Ci]
    unsigned short* g_t     = carve((size_t)B_ * Ci_ * N2_ * 2); // [B,Ci,N2]
    unsigned short* y_b     = carve((size_t)B_ * N_ * Ci_ * 2);  // [B,N,Ci]

    // 0) conversions: x transpose-convert + weight converts
    {
        dim3 tg(N_ / 16, C_ / 16, B_);
        transpose_cvt_kernel<<<tg, dim3(16, 16), 0, stream>>>(x, xt);
        const int nw = Ci_ * C_;
        cvt_f32_bf16_kernel<<<(nw + 255) / 256, 256, 0, stream>>>(w_theta, wq, nw);
        cvt_f32_bf16_kernel<<<(nw + 255) / 256, 256, 0, stream>>>(w_phi,   wk, nw);
        cvt_f32_bf16_kernel<<<(nw + 255) / 256, 256, 0, stream>>>(w_g,     wv, nw);
        cvt_f32_bf16_kernel<<<(nw + 255) / 256, 256, 0, stream>>>(w_final, wfb, nw);
    }

    // 1) QKV projections (WMMA GEMMs, all outputs [N, Cout])
    {
        dim3 grid(N_ / 16, B_);
        conv1x1_gemm_kernel<<<grid, 256, 0, stream>>>(xt, wq, b_theta, theta_t);
        conv1x1_gemm_kernel<<<grid, 256, 0, stream>>>(xt, wk, b_phi,   phi_f);
        conv1x1_gemm_kernel<<<grid, 256, 0, stream>>>(xt, wv, b_g,     g_f);
    }

    // 2) fused 1x2x2 max-pool: phi -> [N2,Ci], g -> transposed [Ci,N2]
    {
        const int np = B_ * N2_ * Ci_;
        pool_1x2x2_kernel<<<(np + 255) / 256, 256, 0, stream>>>(phi_f, g_f,
                                                                phi_p, g_t);
    }

    // 3) flash attention (softmax(theta^T phi) @ g^T), async-LDS pipelined
    {
        dim3 grid(N_ / (16 * 4), B_);
        attention_kernel<<<grid, 128, 0, stream>>>(theta_t, phi_p, g_t, y_b);
    }

    // 4) final conv + BN + residual
    {
        dim3 grid(N_ / 16, B_);
        final_gemm_kernel<<<grid, 256, 0, stream>>>(y_b, wfb, b_final, bn_gamma,
                                                    bn_beta, bn_mean, bn_var, x, out);
    }
}
